// CapsuleMappingTiny_32263794328371
// MI455X (gfx1250) — matrix operand order repaired and checked by reading, not compiled, not run on hardware
//
#include <hip/hip_runtime.h>
#include <hip/hip_bf16.h>

// ---------------------------------------------------------------------------
// CapsuleMappingTiny fused kernel for gfx1250 (MI455X).
// B=32, N=1024, D=256.
//   kn = l2norm(k); qn = l2norm(q)
//   m  = kn @ qn^T               (per batch, 1024x1024)
//   m  = LN_q(m)*gamma1+beta1    -> folded algebraically into GEMM2
//   out= LN_d(m @ v)*gamma2+beta2
// Fused flash-style: never materialize m in HBM.
// ---------------------------------------------------------------------------

typedef __attribute__((ext_vector_type(16))) __bf16        v16bf;
typedef __attribute__((ext_vector_type(8)))  float         v8f;
typedef __attribute__((ext_vector_type(8)))  unsigned int  v8u;

#define NB   32
#define NN   1024
#define DD   256
#define KT   64      // k rows per block
#define QC   64      // q chunk
#define KST  264     // kn/qn LDS row stride (bf16): 528B = 33*16B, conflict-free b128
#define VST  72      // vgT LDS row stride (bf16):  144B =  9*16B
#define MST  72      // m   LDS row stride (bf16)

__device__ __forceinline__ unsigned short f2bf(float x) {
  union { float f; unsigned int u; } c; c.f = x;
  unsigned int u = c.u;
  return (unsigned short)((u + 0x7FFFu + ((u >> 16) & 1u)) >> 16);
}
__device__ __forceinline__ unsigned int pack2bf(float a, float b) {
#if __has_builtin(__builtin_amdgcn_cvt_pk_bf16_f32)
  typedef __attribute__((ext_vector_type(2))) __bf16 v2bf;
  v2bf t = __builtin_amdgcn_cvt_pk_bf16_f32(a, b);
  return __builtin_bit_cast(unsigned int, t);
#else
  return (unsigned int)f2bf(a) | ((unsigned int)f2bf(b) << 16);
#endif
}
// A-operand fragment: two 16-byte runs (K runs of 8 bf16) per lane -> 2x ds_load_b128
__device__ __forceinline__ v16bf frag_pair(const unsigned short* p0, const unsigned short* p1) {
  uint4 a = *(const uint4*)p0;
  uint4 b = *(const uint4*)p1;
  v8u r;
  r[0] = a.x; r[1] = a.y; r[2] = a.z; r[3] = a.w;
  r[4] = b.x; r[5] = b.y; r[6] = b.z; r[7] = b.w;
  return __builtin_bit_cast(v16bf, r);
}
// B-operand fragment: one contiguous 32-byte run (16 bf16) per lane -> 2x ds_load_b128
__device__ __forceinline__ v16bf frag_row16(const unsigned short* p) {
  uint4 a = ((const uint4*)p)[0];
  uint4 b = ((const uint4*)p)[1];
  v8u r;
  r[0] = a.x; r[1] = a.y; r[2] = a.z; r[3] = a.w;
  r[4] = b.x; r[5] = b.y; r[6] = b.z; r[7] = b.w;
  return __builtin_bit_cast(v16bf, r);
}
#define WMMA_BF16(A, Bm, C) \
  __builtin_amdgcn_wmma_f32_16x16x32_bf16(false, (A), false, (Bm), (short)0, (C), false, false)

// ------------------------- kernel 1: row inverse L2 norms -------------------
__global__ __launch_bounds__(256)
void rownorm_kernel(const float* __restrict__ Kg, const float* __restrict__ Qg,
                    float* __restrict__ rnk, float* __restrict__ rnq) {
  const int lane = threadIdx.x & 31;
  const int wave = threadIdx.x >> 5;
  const long row = (long)blockIdx.x * 8 + wave;            // 0 .. B*N-1
  const float* src = (blockIdx.y == 0) ? Kg : Qg;
  float* dst       = (blockIdx.y == 0) ? rnk : rnq;
  const float* p = src + row * DD + lane * 8;
  float4 a = *(const float4*)p;
  float4 b = *(const float4*)(p + 4);
  float s = a.x*a.x + a.y*a.y + a.z*a.z + a.w*a.w
          + b.x*b.x + b.y*b.y + b.z*b.z + b.w*b.w;
  s += __shfl_xor(s, 1, 32);  s += __shfl_xor(s, 2, 32);
  s += __shfl_xor(s, 4, 32);  s += __shfl_xor(s, 8, 32);
  s += __shfl_xor(s, 16, 32);
  if (lane == 0) dst[row] = rsqrtf(fmaxf(s, 1e-12f));
}

// ------------------ kernel 2: G[b,d]=sum_q g1*v, Bv[b,d]=sum_q b1*v ---------
__global__ __launch_bounds__(256)
void vreduce_kernel(const float* __restrict__ Vg, const float* __restrict__ gamma1,
                    const float* __restrict__ beta1,
                    float* __restrict__ Gv, float* __restrict__ Bv) {
  __shared__ float g1[NN], b1[NN];
  for (int i = threadIdx.x; i < NN; i += 256) { g1[i] = gamma1[i]; b1[i] = beta1[i]; }
  __syncthreads();
  const int b = blockIdx.x, d = threadIdx.x;
  const float* vb = Vg + (long)b * NN * DD + d;
  float gs = 0.f, bs = 0.f;
  for (int q = 0; q < NN; ++q) {
    float val = vb[(long)q * DD];
    gs += g1[q] * val; bs += b1[q] * val;
  }
  Gv[b * DD + d] = gs; Bv[b * DD + d] = bs;
}

// ------------------------- kernel 3: fused main -----------------------------
// grid = (N/KT, B), block = 256 threads (8 wave32s)
#define SMEM_TILE_BYTES ((KT*KST + QC*KST + DD*VST + KT*MST) * 2)
#define SMEM_STAT_BYTES ((KT + KT + DD + DD + DD + DD) * 4)
#define SMEM_BYTES      (SMEM_TILE_BYTES + SMEM_STAT_BYTES)

__global__ __launch_bounds__(256)
void fused_capsule_kernel(const float* __restrict__ Kg, const float* __restrict__ Qg,
                          const float* __restrict__ Vg,
                          const float* __restrict__ gamma1,
                          const float* __restrict__ gamma2, const float* __restrict__ beta2,
                          const float* __restrict__ rnk, const float* __restrict__ rnq,
                          const float* __restrict__ Gv, const float* __restrict__ Bvv,
                          float* __restrict__ Out) {
  extern __shared__ unsigned char smem[];
  unsigned short* knL = (unsigned short*)smem;     // [KT][KST]  bf16, kn tile
  unsigned short* qnL = knL + KT * KST;            // [QC][KST]  bf16, qn chunk
  unsigned short* vgL = qnL + QC * KST;            // [DD][VST]  bf16, (gamma1*v)^T chunk
  unsigned short* mL  = vgL + DD * VST;            // [KT][MST]  bf16, m chunk
  float* oL = (float*)smem;                        // [KT][DD]   f32  (aliases tiles, post-loop)
  float* rowsum = (float*)(smem + SMEM_TILE_BYTES);
  float* rowsq  = rowsum + KT;
  float* GL  = rowsq + KT;
  float* BvL = GL + DD;
  float* g2L = BvL + DD;
  float* b2L = g2L + DD;

  const int tid  = threadIdx.x;
  const int b    = blockIdx.y;
  const int kbase = blockIdx.x * KT;
  const int wave = tid >> 5;
  const int lane = tid & 31;
  const int half = lane >> 4;       // 0/1 : lanes 0-15 / 16-31
  const int lr   = lane & 15;

  // stage per-batch vectors, LN2 params, zero row stats
  GL[tid]  = Gv[b * DD + tid];
  BvL[tid] = Bvv[b * DD + tid];
  g2L[tid] = gamma2[tid];
  b2L[tid] = beta2[tid];
  if (tid < KT) { rowsum[tid] = 0.f; rowsq[tid] = 0.f; }

  // load kn tile -> bf16 LDS (row-major, stride KST)
  {
    const float* kb = Kg + ((long)b * NN + kbase) * DD;
    const float* rk = rnk + (long)b * NN + kbase;
    unsigned int* knU = (unsigned int*)knL;
    for (int idx = tid; idx < KT * (DD / 4); idx += 256) {
      int r  = idx >> 6;
      int c4 = (idx & 63) << 2;
      float4 x = *(const float4*)(kb + (long)r * DD + c4);
      float s = rk[r];
      knU[r * (KST / 2) + (c4 >> 1)    ] = pack2bf(x.x * s, x.y * s);
      knU[r * (KST / 2) + (c4 >> 1) + 1] = pack2bf(x.z * s, x.w * s);
    }
  }

  // GEMM2 ownership: wave covers 2 M-tiles x 64 cols (4 N-tiles); B reused 2x.
  const int mtp = wave >> 2;          // 0/1 -> M tiles {2*mtp, 2*mtp+1}
  const int dq  = wave & 3;           // 64-col quad
  const int dqb = dq * 64;
  v8f S[2][4];
  const v8f vzero = {0.f,0.f,0.f,0.f,0.f,0.f,0.f,0.f};
#pragma unroll
  for (int ai = 0; ai < 2; ++ai)
#pragma unroll
    for (int nt = 0; nt < 4; ++nt) S[ai][nt] = vzero;

  // GEMM1 tile assignment: wave w computes m tiles 2w and 2w+1 (same mt)
  const int mtA  = (wave * 2) >> 2;
  const int qtA0 = (wave * 2) & 3;
  const int qtA1 = (wave * 2 + 1) & 3;

  for (int qb = 0; qb < NN; qb += QC) {
    __syncthreads();   // protect qnL/vgL/mL reuse against previous GEMM2

    // --- stage qn chunk (row-major bf16) and vg chunk (transposed bf16) ---
    {
      const float* qsrc = Qg + ((long)b * NN + qb) * DD;
      const float* rq   = rnq + (long)b * NN + qb;
      unsigned int* qnU = (unsigned int*)qnL;
      for (int idx = tid; idx < QC * (DD / 4); idx += 256) {
        int r = idx >> 6;  int c4 = (idx & 63) << 2;
        float4 x = *(const float4*)(qsrc + (long)r * DD + c4);
        float s = rq[r];
        qnU[r * (KST / 2) + (c4 >> 1)    ] = pack2bf(x.x * s, x.y * s);
        qnU[r * (KST / 2) + (c4 >> 1) + 1] = pack2bf(x.z * s, x.w * s);
      }
      // vgT: two q rows per thread, packed b32 stores (q pairs along rows of vgT)
      const float* vsrc = Vg + ((long)b * NN + qb) * DD;
      unsigned int* vgU = (unsigned int*)vgL;
      for (int idx = tid; idx < (QC / 2) * (DD / 4); idx += 256) {
        int qq = idx >> 6;                 // q pair 0..31
        int c4 = (idx & 63) << 2;          // d col group
        const float* r0 = vsrc + (long)(2 * qq) * DD + c4;
        float4 x = *(const float4*)r0;
        float4 y = *(const float4*)(r0 + DD);
        float g0 = gamma1[qb + 2 * qq];
        float g1 = gamma1[qb + 2 * qq + 1];
        vgU[(c4 + 0) * (VST / 2) + qq] = pack2bf(x.x * g0, y.x * g1);
        vgU[(c4 + 1) * (VST / 2) + qq] = pack2bf(x.y * g0, y.y * g1);
        vgU[(c4 + 2) * (VST / 2) + qq] = pack2bf(x.z * g0, y.z * g1);
        vgU[(c4 + 3) * (VST / 2) + qq] = pack2bf(x.w * g0, y.w * g1);
      }
      // prefetch next chunk (global_prefetch_b8)
      if (qb + QC < NN) {
        const char* qn_next = (const char*)(Qg + ((long)b * NN + qb + QC) * DD);
        const char* v_next  = (const char*)(Vg + ((long)b * NN + qb + QC) * DD);
        __builtin_prefetch(qn_next + tid * 256, 0, 1);
        __builtin_prefetch(v_next  + tid * 256, 0, 1);
      }
    }
    __syncthreads();

    // --- GEMM1: m[64x64] chunk = kn_tile @ qn_chunk^T, 2 tiles per wave ---
    // software pipelined across K steps
    v8f C0 = vzero, C1 = vzero;
    {
      const unsigned short* arow = knL + (mtA  * 16 + lr) * KST;
      const unsigned short* br0  = qnL + (qtA0 * 16 + lr) * KST;
      const unsigned short* br1  = qnL + (qtA1 * 16 + lr) * KST;
      v16bf Ac  = frag_pair(arow + 16 * half, arow + 8 + 16 * half);
      v16bf B0c = frag_row16(br0 + 16 * half);
      v16bf B1c = frag_row16(br1 + 16 * half);
#pragma unroll
      for (int dk = 0; dk < DD; dk += 32) {
        v16bf An = Ac, B0n = B0c, B1n = B1c;
        if (dk + 32 < DD) {
          An  = frag_pair(arow + dk + 32 + 16 * half, arow + dk + 40 + 16 * half);
          B0n = frag_row16(br0 + dk + 32 + 16 * half);
          B1n = frag_row16(br1 + dk + 32 + 16 * half);
        }
        C0 = WMMA_BF16(Ac, B0c, C0);
        C1 = WMMA_BF16(Ac, B1c, C1);
        Ac = An; B0c = B0n; B1c = B1n;
      }
    }

    // --- row stats (sum, sum^2 over q) + spill m to LDS as bf16 ---
#pragma unroll
    for (int g = 0; g < 8; ++g) {
      float v0 = C0[g], v1 = C1[g];
      float s  = v0 + v1;
      float sq = v0 * v0 + v1 * v1;
      s  += __shfl_xor(s, 1, 16);  s  += __shfl_xor(s, 2, 16);
      s  += __shfl_xor(s, 4, 16);  s  += __shfl_xor(s, 8, 16);
      sq += __shfl_xor(sq, 1, 16); sq += __shfl_xor(sq, 2, 16);
      sq += __shfl_xor(sq, 4, 16); sq += __shfl_xor(sq, 8, 16);
      int row = mtA * 16 + g + half * 8;
      if (lr == 0) { atomicAdd(&rowsum[row], s); atomicAdd(&rowsq[row], sq); }
      mL[row * MST + qtA0 * 16 + lr] = f2bf(v0);
      mL[row * MST + qtA1 * 16 + lr] = f2bf(v1);
    }
    __syncthreads();

    // --- GEMM2: S[2][4] += m_tiles @ vg ; B pair-pipelined, reused 2x ---
    {
      const unsigned short* mRow0 = mL + (mtp * 32 + lr) * MST;
      const unsigned short* mRow1 = mRow0 + 16 * MST;
#pragma unroll
      for (int kk = 0; kk < QC; kk += 32) {
        v16bf A0 = frag_pair(mRow0 + kk + 16 * half, mRow0 + kk + 8 + 16 * half);
        v16bf A1 = frag_pair(mRow1 + kk + 16 * half, mRow1 + kk + 8 + 16 * half);
        const unsigned short* vgb = vgL + (dqb + lr) * VST + kk + 16 * half;
        v16bf Ba = frag_row16(vgb);
        v16bf Bb = frag_row16(vgb + 16 * VST);
#pragma unroll
        for (int nt = 0; nt < 4; nt += 2) {
          v16bf Bc = Ba, Bd = Bb;
          if (nt + 2 < 4) {
            Bc = frag_row16(vgb + (nt + 2) * 16 * VST);
            Bd = frag_row16(vgb + (nt + 3) * 16 * VST);
          }
          S[0][nt]     = WMMA_BF16(A0, Ba, S[0][nt]);
          S[1][nt]     = WMMA_BF16(A1, Ba, S[1][nt]);
          S[0][nt + 1] = WMMA_BF16(A0, Bb, S[0][nt + 1]);
          S[1][nt + 1] = WMMA_BF16(A1, Bb, S[1][nt + 1]);
          Ba = Bc; Bb = Bd;
        }
      }
    }
  }
  __syncthreads();

  // --- apply folded LN1: o = rs*S - rs*mu*G + Bv ; stash in LDS (aliased) ---
#pragma unroll
  for (int ai = 0; ai < 2; ++ai) {
#pragma unroll
    for (int g = 0; g < 8; ++g) {
      int row = (mtp * 2 + ai) * 16 + g + half * 8;
      float mu  = rowsum[row] * (1.f / (float)NN);
      float var = rowsq[row] * (1.f / (float)NN) - mu * mu;
      float rs  = rsqrtf(var + 1e-3f);
#pragma unroll
      for (int nt = 0; nt < 4; ++nt) {
        int dcol = dqb + nt * 16 + lr;
        float o = rs * S[ai][nt][g] - rs * mu * GL[dcol] + BvL[dcol];
        oL[row * DD + dcol] = o;
      }
    }
  }
  __syncthreads();

  // --- LN2 over D=256 per row, write output (wave handles 8 rows) ---
  for (int rr = 0; rr < 8; ++rr) {
    int row = wave * 8 + rr;
    const float* orow = oL + row * DD + lane * 8;
    float4 x0 = *(const float4*)orow;
    float4 x1 = *(const float4*)(orow + 4);
    float s  = x0.x + x0.y + x0.z + x0.w + x1.x + x1.y + x1.z + x1.w;
    float sq = x0.x*x0.x + x0.y*x0.y + x0.z*x0.z + x0.w*x0.w
             + x1.x*x1.x + x1.y*x1.y + x1.z*x1.z + x1.w*x1.w;
    s  += __shfl_xor(s, 1, 32);  s  += __shfl_xor(s, 2, 32);
    s  += __shfl_xor(s, 4, 32);  s  += __shfl_xor(s, 8, 32);  s  += __shfl_xor(s, 16, 32);
    sq += __shfl_xor(sq, 1, 32); sq += __shfl_xor(sq, 2, 32);
    sq += __shfl_xor(sq, 4, 32); sq += __shfl_xor(sq, 8, 32); sq += __shfl_xor(sq, 16, 32);
    float mu  = s * (1.f / (float)DD);
    float var = sq * (1.f / (float)DD) - mu * mu;
    float rs  = rsqrtf(var + 1e-3f);
    int d0 = lane * 8;
    float4 y0, y1;
    y0.x = (x0.x - mu) * rs * g2L[d0 + 0] + b2L[d0 + 0];
    y0.y = (x0.y - mu) * rs * g2L[d0 + 1] + b2L[d0 + 1];
    y0.z = (x0.z - mu) * rs * g2L[d0 + 2] + b2L[d0 + 2];
    y0.w = (x0.w - mu) * rs * g2L[d0 + 3] + b2L[d0 + 3];
    y1.x = (x1.x - mu) * rs * g2L[d0 + 4] + b2L[d0 + 4];
    y1.y = (x1.y - mu) * rs * g2L[d0 + 5] + b2L[d0 + 5];
    y1.z = (x1.z - mu) * rs * g2L[d0 + 6] + b2L[d0 + 6];
    y1.w = (x1.w - mu) * rs * g2L[d0 + 7] + b2L[d0 + 7];
    float* op = Out + ((long)b * NN + kbase + row) * DD + d0;
    *(float4*)op       = y0;
    *(float4*)(op + 4) = y1;
  }
}

// ------------------------------- launcher -----------------------------------
extern "C" void kernel_launch(void* const* d_in, const int* in_sizes, int n_in,
                              void* d_out, int out_size, void* d_ws, size_t ws_size,
                              hipStream_t stream) {
  (void)in_sizes; (void)n_in; (void)out_size; (void)ws_size;
  const float* k      = (const float*)d_in[0];
  const float* q      = (const float*)d_in[1];
  const float* v      = (const float*)d_in[2];
  const float* gamma1 = (const float*)d_in[3];
  const float* beta1  = (const float*)d_in[4];
  const float* gamma2 = (const float*)d_in[5];
  const float* beta2  = (const float*)d_in[6];
  float* out = (float*)d_out;
  float* ws  = (float*)d_ws;
  float* rnk = ws;                    // [B*N]
  float* rnq = ws + NB * NN;          // [B*N]
  float* Gv  = ws + 2 * NB * NN;      // [B*D]
  float* Bv  = Gv + NB * DD;          // [B*D]

  rownorm_kernel<<<dim3((NB * NN) / 8, 2), 256, 0, stream>>>(k, q, rnk, rnq);
  vreduce_kernel<<<NB, 256, 0, stream>>>(v, gamma1, beta1, Gv, Bv);

  (void)hipFuncSetAttribute((const void*)fused_capsule_kernel,
                            hipFuncAttributeMaxDynamicSharedMemorySize, SMEM_BYTES);
  fused_capsule_kernel<<<dim3(NN / KT, NB), 256, SMEM_BYTES, stream>>>(
      k, q, v, gamma1, gamma2, beta2, rnk, rnq, Gv, Bv, out);
}
